// contextual_attention_block_19722489823323
// MI455X (gfx1250) — compile-verified, matrix operand order checked by reading, not run on hardware
//
#include <hip/hip_runtime.h>
#include <hip/hip_bf16.h>

// ---------------------------------------------------------------------------
// Contextual-attention block for MI455X (gfx1250, wave32, WMMA).
//
// All matrix-shaped work (attention scores 1024x1024x1152, transposed-conv
// 1024x2048x1024, both 3x3 convs via im2col 16384x128x1152, both pointwise
// convs) runs through a single generic batched bf16 WMMA GEMM
// (v_wmma_f32_16x16x32_bf16, f32 accumulate) with DOUBLE-BUFFERED
// ASYNC global->LDS staging (global_load_async_to_lds_b128 + s_wait_asynccnt)
// so HBM tile fetch overlaps the matrix math. Elementwise / stencil /
// softmax / pooling stages are bandwidth-bound VALU kernels.
//
// Workspace: statically laid out with lifetime-based aliasing.
// Peak requirement: ~110.2 MB (WS_NEED below).
// ---------------------------------------------------------------------------

typedef __attribute__((ext_vector_type(16))) __bf16 v16bf;
typedef __attribute__((ext_vector_type(8)))  float  v8f;

// Async 16-byte global->LDS copy (per-lane), tracked by ASYNCcnt.
// VDST = VGPR with LDS byte address (low 32 bits of the generic LDS pointer,
// per the aperture mapping in ISA 10.2), VADDR = 64-bit global address.
static __device__ __forceinline__ void async_cp16(const void* gsrc, void* lds_dst)
{
    const unsigned long long ga  = (unsigned long long)gsrc;
    const unsigned int       lof = (unsigned int)(unsigned long long)lds_dst;
    asm volatile("global_load_async_to_lds_b128 %0, %1, off"
                 :: "v"(lof), "v"(ga)
                 : "memory");
}

// ------------------------------ GEMM (WMMA) --------------------------------
// C[m,n] = sum_k A[m,k] * B[n,k]   (A: MxK bf16, B: NxK bf16, C: MxN f32)
// epilogue: v = acc * colScale[n] + colBias[n]; if accum: v += C_old.
// M % 64 == 0, N % 64 == 0, K % 32 == 0 (guaranteed by all call sites).
__global__ void gemm_bf16_wmma(const __bf16* __restrict__ A,
                               const __bf16* __restrict__ Bm,
                               float* __restrict__ Cm,
                               int M, int N, int K,
                               int lda, int ldb, int ldc,
                               long long sA, long long sB, long long sC,
                               const float* __restrict__ colScale, long long sScale,
                               const float* __restrict__ colBias, int accum)
{
    // double-buffered tiles: 64 rows x 32 K, row stride 40 (bank pad)
    __shared__ __bf16 As[2][64 * 40];
    __shared__ __bf16 Bs[2][64 * 40];

    const int tid  = threadIdx.x;          // 128 threads = 4 waves
    const int lane = tid & 31;
    const int wv   = tid >> 5;
    const int bz   = blockIdx.z;
    const int m0   = blockIdx.y * 64;
    const int n0   = blockIdx.x * 64;

    A  += (long long)bz * sA;
    Bm += (long long)bz * sB;
    Cm += (long long)bz * sC;

    const int wm = (wv & 1) * 32;          // wave's 32x32 quadrant
    const int wn = (wv >> 1) * 32;

    v8f acc[2][2];
#pragma unroll
    for (int i = 0; i < 2; ++i)
#pragma unroll
        for (int j = 0; j < 2; ++j)
#pragma unroll
            for (int e = 0; e < 8; ++e) acc[i][j][e] = 0.0f;

    const int r     = tid >> 1;            // 0..63  (tile row staged by this thread)
    const int half  = tid & 1;             // which 16-element half of the 32-K row
    const int stoff = r * 40 + half * 16;  // element offset inside a tile buffer

    // issue the 4 async 16B copies (2 for A, 2 for B) of one 64x32 tile pair
    auto issue_tile = [&](int buf, int k0) {
        const __bf16* ga = A  + (long long)(m0 + r) * lda + (k0 + half * 16);
        const __bf16* gb = Bm + (long long)(n0 + r) * ldb + (k0 + half * 16);
        async_cp16(ga,     &As[buf][stoff]);
        async_cp16(ga + 8, &As[buf][stoff + 8]);
        async_cp16(gb,     &Bs[buf][stoff]);
        async_cp16(gb + 8, &Bs[buf][stoff + 8]);
    };

    const int nk = K >> 5;
    issue_tile(0, 0);

    for (int ks = 0; ks < nk; ++ks) {
        const int cur = ks & 1;
        if (ks + 1 < nk) {
            issue_tile(cur ^ 1, (ks + 1) << 5);   // prefetch next tile
            // async loads complete in order: <=4 outstanding => tile ks landed
            asm volatile("s_wait_asynccnt 0x4" ::: "memory");
        } else {
            asm volatile("s_wait_asynccnt 0x0" ::: "memory");
        }
        __syncthreads();   // publish this tile's LDS writes to all 4 waves

        // ISA 16-bit fragment layout: lane l -> row (l&15),
        // K = {kb..kb+7, kb+16..kb+23}, kb = (l>>4)*8.
        const int kb = (lane >> 4) * 8;
        const int rl = lane & 15;
        union FragU { v16bf v; uint4 u[2]; } fa[2], fb[2];
#pragma unroll
        for (int t = 0; t < 2; ++t) {
            const int mr = wm + t * 16 + rl;
            fa[t].u[0] = *reinterpret_cast<const uint4*>(&As[cur][mr * 40 + kb]);
            fa[t].u[1] = *reinterpret_cast<const uint4*>(&As[cur][mr * 40 + kb + 16]);
            const int nr = wn + t * 16 + rl;
            fb[t].u[0] = *reinterpret_cast<const uint4*>(&Bs[cur][nr * 40 + kb]);
            fb[t].u[1] = *reinterpret_cast<const uint4*>(&Bs[cur][nr * 40 + kb + 16]);
        }
#pragma unroll
        for (int mt = 0; mt < 2; ++mt)
#pragma unroll
            for (int nt = 0; nt < 2; ++nt)
                acc[mt][nt] = __builtin_amdgcn_wmma_f32_16x16x32_bf16(
                    false, fa[mt].v, false, fb[nt].v,
                    (short)0, acc[mt][nt], false, false);

        __syncthreads();   // everyone done reading buf `cur` before it's refilled
    }

    // C/D layout: lane l -> col (l&15), rows (l>>4)*8 + e.
    const int cl    = lane & 15;
    const int rbase = (lane >> 4) * 8;
    for (int mt = 0; mt < 2; ++mt) {
        for (int nt = 0; nt < 2; ++nt) {
            const int n   = n0 + wn + nt * 16 + cl;
            const float s = colScale ? colScale[(long long)bz * sScale + n] : 1.0f;
            const float b = colBias ? colBias[n] : 0.0f;
#pragma unroll
            for (int e = 0; e < 8; ++e) {
                const int m = m0 + wm + mt * 16 + rbase + e;
                const long long ci = (long long)m * ldc + n;
                float v = acc[mt][nt][e] * s + b;
                if (accum) v += Cm[ci];
                Cm[ci] = v;
            }
        }
    }
}

// --------------------- attention-score preparation -------------------------
// P[b,p, t*128+c] = 3x3xC patch of x_ds (x[::2,::2]) around p=(i,j), zero-pad.
// norms[b,p] = sum of squares (the w*w normalizer).
__global__ void k_patches(const float* __restrict__ x, __bf16* __restrict__ P,
                          float* __restrict__ norms)
{
    __shared__ float red[128];
    const int bp = blockIdx.x;             // b*1024 + p
    const int b  = bp >> 10;
    const int p  = bp & 1023;
    const int i  = p >> 5, j = p & 31;
    const int c  = threadIdx.x;            // 0..127
    float ss = 0.f;
#pragma unroll
    for (int t = 0; t < 9; ++t) {
        const int di = t / 3, dj = t % 3;
        const int hh = i + di - 1, ww = j + dj - 1;
        float v = 0.f;
        if (hh >= 0 && hh < 32 && ww >= 0 && ww < 32)
            v = x[(((long long)b * 64 + hh * 2) * 64 + ww * 2) * 128 + c];
        P[((long long)bp * 9 + t) * 128 + c] = (__bf16)v;
        ss += v * v;
    }
    red[c] = ss; __syncthreads();
    for (int s = 64; s > 0; s >>= 1) {
        if (c < s) red[c] += red[c + s];
        __syncthreads();
    }
    if (c == 0) norms[bp] = red[0];
}

__global__ void k_invnorm(const float* __restrict__ n, float* __restrict__ inv, int cnt)
{
    const int i = blockIdx.x * blockDim.x + threadIdx.x;
    if (i < cnt) inv[i] = 1.0f / fmaxf(n[i], 1e-8f);
}

// ------------------------------- fusion ------------------------------------
__global__ void k_fuse1(const float* __restrict__ Y, float* __restrict__ Z)
{
    const long long idx = (long long)blockIdx.x * blockDim.x + threadIdx.x;
    if (idx >= (long long)4 * 1024 * 1024) return;
    const int q = (int)(idx & 1023);
    const int p = (int)((idx >> 10) & 1023);
    const float* Yb = Y + ((idx >> 20) << 20);
    float s = 0.f;
    for (int dp = -1; dp <= 1; ++dp) {
        const int pp = p + dp;
        if (pp < 0 || pp >= 1024) continue;
        for (int dq = -1; dq <= 1; ++dq) {
            const int qq = q + dq;
            if (qq < 0 || qq >= 1024) continue;
            s += Yb[(pp << 10) + qq];
        }
    }
    Z[idx] = s;
}

__device__ __forceinline__ int permi(int a) { return ((a & 31) << 5) | (a >> 5); }

__global__ void k_fuse2(const float* __restrict__ Z, float* __restrict__ O)
{
    const long long idx = (long long)blockIdx.x * blockDim.x + threadIdx.x;
    if (idx >= (long long)4 * 1024 * 1024) return;
    const int q = (int)(idx & 1023);
    const int p = (int)((idx >> 10) & 1023);
    const float* Zb = Z + ((idx >> 20) << 20);
    const int a0 = permi(p);               // transposed flattening
    const int c0 = permi(q);
    float s = 0.f;
    for (int da = -1; da <= 1; ++da) {
        const int a = a0 + da;
        if (a < 0 || a >= 1024) continue;
        const int pa = permi(a);
        for (int db = -1; db <= 1; ++db) {
            const int cq = c0 + db;
            if (cq < 0 || cq >= 1024) continue;
            s += Zb[(pa << 10) + permi(cq)];
        }
    }
    O[idx] = s;
}

// ---------------------------- mask pipeline --------------------------------
__global__ void k_pool16(const float* __restrict__ mask, float* __restrict__ mp)
{
    const int idx = blockIdx.x * blockDim.x + threadIdx.x;  // B*32*32
    if (idx >= 4 * 1024) return;
    const int b = idx >> 10, i = (idx >> 5) & 31, j = idx & 31;
    const float* mb = mask + (long long)b * 512 * 512;
    float mx = -1e30f;
    for (int u = 0; u < 16; ++u)
        for (int v = 0; v < 16; ++v)
            mx = fmaxf(mx, mb[(i * 16 + u) * 512 + j * 16 + v]);
    mp[idx] = mx;
}

__global__ void k_pool3mm(const float* __restrict__ mp, float* __restrict__ mm)
{
    const int idx = blockIdx.x * blockDim.x + threadIdx.x;  // B*32*32
    if (idx >= 4 * 1024) return;
    const int b = idx >> 10, i = (idx >> 5) & 31, j = idx & 31;
    const float* mb = mp + (b << 10);
    float mx = -1e30f;
    for (int di = -1; di <= 1; ++di) {
        const int ii = i + di; if (ii < 0 || ii >= 32) continue;
        for (int dj = -1; dj <= 1; ++dj) {
            const int jj = j + dj; if (jj < 0 || jj >= 32) continue;
            mx = fmaxf(mx, mb[(ii << 5) + jj]);
        }
    }
    mm[idx] = 1.0f - mx;
}

// ------------------------------ softmax ------------------------------------
// corres[b,p,q] = softmax_q(Y*mm*10) * mm ; writes f32 (output) + bf16 (GEMM A).
__global__ void k_softmax(const float* __restrict__ Y, const float* __restrict__ mm,
                          float* __restrict__ cf, __bf16* __restrict__ cbf)
{
    __shared__ float red[256];
    const int bp  = blockIdx.x;            // b*1024 + p
    const int b   = bp >> 10;
    const int tid = threadIdx.x;
    const float* row = Y + ((long long)bp << 10);
    const float* mb  = mm + (b << 10);
    float sv[4];
    float mx = -1e30f;
#pragma unroll
    for (int t = 0; t < 4; ++t) {
        const int q = tid + t * 256;
        sv[t] = row[q] * mb[q] * 10.0f;
        mx = fmaxf(mx, sv[t]);
    }
    red[tid] = mx; __syncthreads();
    for (int s = 128; s > 0; s >>= 1) {
        if (tid < s) red[tid] = fmaxf(red[tid], red[tid + s]);
        __syncthreads();
    }
    mx = red[0]; __syncthreads();
    float sm = 0.f;
#pragma unroll
    for (int t = 0; t < 4; ++t) { sv[t] = __expf(sv[t] - mx); sm += sv[t]; }
    red[tid] = sm; __syncthreads();
    for (int s = 128; s > 0; s >>= 1) {
        if (tid < s) red[tid] += red[tid + s];
        __syncthreads();
    }
    const float inv = 1.0f / red[0];
#pragma unroll
    for (int t = 0; t < 4; ++t) {
        const int q = tid + t * 256;
        const float v = sv[t] * inv * mb[q];
        cf[((long long)bp << 10) + q]  = v;
        cbf[((long long)bp << 10) + q] = (__bf16)v;
    }
}

// -------------------- transposed-conv (deconv) path ------------------------
// Rwt[b, n=(dh,dw,c), q] = raw 4x4-stride-2 patch values of x (pad 1), bf16,
// stored K-major in q so tmp = corres x Rwt^T fits the generic GEMM.
__global__ void k_build_rwt(const float* __restrict__ x, __bf16* __restrict__ Rwt)
{
    const long long idx = (long long)blockIdx.x * blockDim.x + threadIdx.x;
    if (idx >= (long long)4 * 2048 * 1024) return;
    const int q = (int)(idx & 1023);
    const int n = (int)((idx >> 10) & 2047);
    const int b = (int)(idx >> 21);
    const int c = n & 127;
    const int dh = (n >> 7) >> 2, dw = (n >> 7) & 3;
    const int i = q >> 5, j = q & 31;
    const int hh = 2 * i - 1 + dh, ww = 2 * j - 1 + dw;
    float v = 0.f;
    if (hh >= 0 && hh < 64 && ww >= 0 && ww < 64)
        v = x[(((long long)b * 64 + hh) * 64 + ww) * 128 + c];
    Rwt[idx] = (__bf16)v;
}

// stride-2 overlap-add: y_up[b,h,w,c] = sum over covering patches of tmp.
__global__ void k_overlap(const float* __restrict__ tmp, float* __restrict__ yup)
{
    const long long idx = (long long)blockIdx.x * blockDim.x + threadIdx.x;
    if (idx >= (long long)4 * 64 * 64 * 128) return;
    const int c = (int)(idx & 127);
    const int w = (int)((idx >> 7) & 63);
    const int h = (int)((idx >> 13) & 63);
    const int b = (int)(idx >> 19);
    const float* tb = tmp + (long long)b * 1024 * 2048;
    float s = 0.f;
    const int ph = (h + 1) & 1, pw = (w + 1) & 1;
    for (int u = 0; u < 2; ++u) {
        const int dh = ph + u * 2;
        const int i  = (h + 1 - dh) / 2;
        if (i < 0 || i >= 32) continue;
        for (int v2 = 0; v2 < 2; ++v2) {
            const int dw = pw + v2 * 2;
            const int j  = (w + 1 - dw) / 2;
            if (j < 0 || j >= 32) continue;
            s += tb[((i * 32 + j) << 11) + ((dh * 4 + dw) << 7) + c];
        }
    }
    yup[idx] = s;
}

// ------------------------- conv / gating helpers ---------------------------
// Pack conv weights (3,3,CinTot,128) f32 -> bf16 [co, (t*128+ci)] for a
// 128-wide Cin slice starting at ci_off.
__global__ void k_pack_convw(const float* __restrict__ wsrc, __bf16* __restrict__ dst,
                             int CinTot, int ci_off)
{
    const int idx = blockIdx.x * blockDim.x + threadIdx.x;  // 128*1152
    if (idx >= 128 * 1152) return;
    const int k  = idx % 1152;
    const int co = idx / 1152;
    const int t  = k >> 7, ci = k & 127;
    dst[idx] = (__bf16)wsrc[((long long)t * CinTot + ci_off + ci) * 128 + co];
}

// Pack pointwise weights (Cin,128) f32 -> bf16 [co, ci].
__global__ void k_pack_pw(const float* __restrict__ pw, __bf16* __restrict__ dst, int Cin)
{
    const int idx = blockIdx.x * blockDim.x + threadIdx.x;  // 128*Cin
    if (idx >= 128 * Cin) return;
    const int ci = idx % Cin;
    const int co = idx / Cin;
    dst[idx] = (__bf16)pw[(long long)ci * 128 + co];
}

// im2col (3x3 SAME, Cin=128, f32 src (B,64,64,128)) -> bf16 [16384, 1152].
__global__ void k_im2col(const float* __restrict__ src, __bf16* __restrict__ col)
{
    const long long idx = (long long)blockIdx.x * blockDim.x + threadIdx.x;
    if (idx >= (long long)16384 * 1152) return;
    const int k = (int)(idx % 1152);
    const long long m = idx / 1152;
    const int w = (int)(m & 63), h = (int)((m >> 6) & 63), b = (int)(m >> 12);
    const int t = k >> 7, ci = k & 127;
    const int hh = h + t / 3 - 1, ww = w + t % 3 - 1;
    float v = 0.f;
    if (hh >= 0 && hh < 64 && ww >= 0 && ww < 64)
        v = src[(((long long)b * 64 + hh) * 64 + ww) * 128 + ci];
    col[idx] = (__bf16)v;
}

// Depthwise 3x3 SAME on an implicit concat [s0 (C0 ch) | s1 (Ct-C0 ch)].
__global__ void k_depthwise(const float* __restrict__ s0, const float* __restrict__ s1,
                            int C0, int Ct, const float* __restrict__ dwW,
                            float* __restrict__ out)
{
    const long long idx = (long long)blockIdx.x * blockDim.x + threadIdx.x;
    if (idx >= (long long)4 * 4096 * Ct) return;
    const int c = (int)(idx % Ct);
    const long long m = idx / Ct;
    const int w = (int)(m & 63), h = (int)((m >> 6) & 63), b = (int)(m >> 12);
    const float* src = (c < C0) ? s0 : s1;
    const int cl = (c < C0) ? c : (c - C0);
    float s = 0.f;
    for (int di = -1; di <= 1; ++di) {
        const int hh = h + di; if (hh < 0 || hh >= 64) continue;
        for (int dj = -1; dj <= 1; ++dj) {
            const int ww = w + dj; if (ww < 0 || ww >= 64) continue;
            s += src[(((long long)b * 64 + hh) * 64 + ww) * 128 + cl] *
                 dwW[((di + 1) * 3 + (dj + 1)) * Ct + c];
        }
    }
    out[idx] = s;
}

__global__ void k_cvt_bf16(const float* __restrict__ src, __bf16* __restrict__ dst,
                           long long n)
{
    const long long i = (long long)blockIdx.x * blockDim.x + threadIdx.x;
    if (i < n) dst[i] = (__bf16)src[i];
}

// out[m,c] = elu(x1raw[m,c]) * softmax_c(x2raw[m,:])[c], C=128.
__global__ void k_gate(const float* __restrict__ x1raw, const float* __restrict__ x2raw,
                       float* __restrict__ out)
{
    __shared__ float red[128];
    const long long m = blockIdx.x;        // 16384 rows
    const int c = threadIdx.x;             // 128
    const float a = x1raw[m * 128 + c];
    const float e = a > 0.f ? a : (__expf(a) - 1.f);
    const float g = x2raw[m * 128 + c];
    red[c] = g; __syncthreads();
    for (int s = 64; s > 0; s >>= 1) {
        if (c < s) red[c] = fmaxf(red[c], red[c + s]);
        __syncthreads();
    }
    const float mx = red[0]; __syncthreads();
    const float ex = __expf(g - mx);
    red[c] = ex; __syncthreads();
    for (int s = 64; s > 0; s >>= 1) {
        if (c < s) red[c] += red[c + s];
        __syncthreads();
    }
    out[m * 128 + c] = e * (ex / red[0]);
}

// ------------------------- workspace layout (bytes) ------------------------
static constexpr size_t OFF_P     = 0;                       //  9,437,184  P bf16
static constexpr size_t OFF_NORM  = 9437184;                 //     16,384
static constexpr size_t OFF_INVN  = OFF_NORM  + 16384;       //     16,384
static constexpr size_t OFF_MPOOL = OFF_INVN  + 16384;       //     16,384
static constexpr size_t OFF_MM    = OFF_MPOOL + 16384;       //     16,384
static constexpr size_t OFF_Y0    = OFF_MM    + 16384;       // 16,777,216
static constexpr size_t OFF_Y1    = OFF_Y0    + 16777216;    // 16,777,216
static constexpr size_t OFF_CORBF = OFF_Y1    + 16777216;    //  8,388,608
static constexpr size_t OFF_RWT   = OFF_CORBF + 8388608;     // 16,777,216
static constexpr size_t OFF_TMP   = OFF_RWT   + 16777216;    // 33,554,432
static constexpr size_t OFF_YUP   = OFF_TMP   + 33554432;    //  8,388,608
static constexpr size_t WS_NEED   = OFF_YUP   + 8388608;     // 110,166,016

// Phase-2 aliases (underlying regions are dead by the time they're reused).
static constexpr size_t OFF_COL    = 0;                      // 37,748,736 (over P..Y1 head)
static constexpr size_t OFF_W1M    = OFF_COL + 37748736;     //    294,912 (inside dead Y1)
static constexpr size_t OFF_W2AM   = OFF_W1M  + 294912;
static constexpr size_t OFF_W2BM   = OFF_W2AM + 294912;
static constexpr size_t OFF_PW1M   = OFF_W2BM + 294912;      //     32,768
static constexpr size_t OFF_PW2M   = OFF_PW1M + 32768;       //     65,536
static constexpr size_t OFF_C1RAW  = OFF_CORBF;              //  8,388,608
static constexpr size_t OFF_DW1    = OFF_RWT;                //  8,388,608
static constexpr size_t OFF_PWIN1  = OFF_RWT + 8388608;      //  4,194,304
static constexpr size_t OFF_PW1OUT = OFF_TMP;                //  8,388,608
static constexpr size_t OFF_G1     = OFF_TMP + 8388608;      //  8,388,608
static constexpr size_t OFF_C2RAW  = OFF_TMP + 16777216;     //  8,388,608
static constexpr size_t OFF_DW2    = OFF_TMP + 25165824;     // 16,777,216 (over dead YUP)
static constexpr size_t OFF_PWIN2  = 0;                      //  8,388,608 (over dead COL)
static constexpr size_t OFF_PW2OUT = 8388608;                //  8,388,608

static inline unsigned cdiv(long long a, long long b) { return (unsigned)((a + b - 1) / b); }

extern "C" void kernel_launch(void* const* d_in, const int* in_sizes, int n_in,
                              void* d_out, int out_size, void* d_ws, size_t ws_size,
                              hipStream_t stream)
{
    (void)in_sizes; (void)n_in; (void)out_size; (void)ws_size;

    const float* x    = (const float*)d_in[0];
    const float* mask = (const float*)d_in[1];
    const float* w1   = (const float*)d_in[2];
    const float* b1   = (const float*)d_in[3];
    const float* dw1  = (const float*)d_in[4];
    const float* pw1  = (const float*)d_in[5];
    const float* pb1  = (const float*)d_in[6];
    const float* w2   = (const float*)d_in[7];
    const float* b2   = (const float*)d_in[8];
    const float* dw2  = (const float*)d_in[9];
    const float* pw2  = (const float*)d_in[10];
    const float* pb2  = (const float*)d_in[11];

    char* ws = (char*)d_ws;
    __bf16* P      = (__bf16*)(ws + OFF_P);
    float*  norms  = (float*) (ws + OFF_NORM);
    float*  invn   = (float*) (ws + OFF_INVN);
    float*  mpool  = (float*) (ws + OFF_MPOOL);
    float*  mm     = (float*) (ws + OFF_MM);
    float*  Y0     = (float*) (ws + OFF_Y0);
    float*  Y1     = (float*) (ws + OFF_Y1);
    __bf16* corbf  = (__bf16*)(ws + OFF_CORBF);
    __bf16* rwt    = (__bf16*)(ws + OFF_RWT);
    float*  tmp    = (float*) (ws + OFF_TMP);
    float*  yup    = (float*) (ws + OFF_YUP);

    __bf16* col    = (__bf16*)(ws + OFF_COL);
    __bf16* w1m    = (__bf16*)(ws + OFF_W1M);
    __bf16* w2am   = (__bf16*)(ws + OFF_W2AM);
    __bf16* w2bm   = (__bf16*)(ws + OFF_W2BM);
    __bf16* pw1m   = (__bf16*)(ws + OFF_PW1M);
    __bf16* pw2m   = (__bf16*)(ws + OFF_PW2M);
    float*  c1raw  = (float*) (ws + OFF_C1RAW);
    float*  dwo1   = (float*) (ws + OFF_DW1);
    __bf16* pwin1  = (__bf16*)(ws + OFF_PWIN1);
    float*  pw1out = (float*) (ws + OFF_PW1OUT);
    float*  g1     = (float*) (ws + OFF_G1);
    float*  c2raw  = (float*) (ws + OFF_C2RAW);
    float*  dwo2   = (float*) (ws + OFF_DW2);
    __bf16* pwin2  = (__bf16*)(ws + OFF_PWIN2);
    float*  pw2out = (float*) (ws + OFF_PW2OUT);

    float* out_f    = (float*)d_out;                 // (4,64,64,128)
    float* corres_f = out_f + 4 * 64 * 64 * 128;     // (4,32,32,1024)

    // ---- phase 1: attention scores ----
    k_patches<<<4096, 128, 0, stream>>>(x, P, norms);
    k_invnorm<<<16, 256, 0, stream>>>(norms, invn, 4096);

    // Y0[b] = (P Pᵀ) * (1/||w_q||²) : M=N=1024, K=1152
    gemm_bf16_wmma<<<dim3(16, 16, 4), 128, 0, stream>>>(
        P, P, Y0, 1024, 1024, 1152, 1152, 1152, 1024,
        (long long)1024 * 1152, (long long)1024 * 1152, (long long)1024 * 1024,
        invn, 1024, nullptr, 0);

    k_fuse1<<<16384, 256, 0, stream>>>(Y0, Y1);
    k_fuse2<<<16384, 256, 0, stream>>>(Y1, Y0);

    k_pool16<<<16, 256, 0, stream>>>(mask, mpool);
    k_pool3mm<<<16, 256, 0, stream>>>(mpool, mm);

    k_softmax<<<4096, 256, 0, stream>>>(Y0, mm, corres_f, corbf);

    // ---- transposed conv: tmp = corres x raw_wᵀ, then overlap-add ----
    k_build_rwt<<<32768, 256, 0, stream>>>(x, rwt);
    gemm_bf16_wmma<<<dim3(32, 16, 4), 128, 0, stream>>>(
        corbf, rwt, tmp, 1024, 2048, 1024, 1024, 1024, 2048,
        (long long)1024 * 1024, (long long)2048 * 1024, (long long)1024 * 2048,
        nullptr, 0, nullptr, 0);
    k_overlap<<<8192, 256, 0, stream>>>(tmp, yup);

    // ---- pack weights (reuses dead Y1 region; must run after softmax) ----
    k_pack_convw<<<cdiv(128 * 1152, 256), 256, 0, stream>>>(w1, w1m, 128, 0);
    k_pack_convw<<<cdiv(128 * 1152, 256), 256, 0, stream>>>(w2, w2am, 256, 0);
    k_pack_convw<<<cdiv(128 * 1152, 256), 256, 0, stream>>>(w2, w2bm, 256, 128);
    k_pack_pw<<<cdiv(128 * 128, 256), 256, 0, stream>>>(pw1, pw1m, 128);
    k_pack_pw<<<cdiv(128 * 256, 256), 256, 0, stream>>>(pw2, pw2m, 256);

    // ---- gated block 1 (input y_up) ----
    k_im2col<<<cdiv((long long)16384 * 1152, 256), 256, 0, stream>>>(yup, col);
    gemm_bf16_wmma<<<dim3(2, 256, 1), 128, 0, stream>>>(
        col, w1m, c1raw, 16384, 128, 1152, 1152, 1152, 128,
        0, 0, 0, nullptr, 0, b1, 0);
    k_depthwise<<<cdiv((long long)4 * 4096 * 128, 256), 256, 0, stream>>>(
        yup, yup, 128, 128, dw1, dwo1);
    k_cvt_bf16<<<cdiv((long long)16384 * 128, 256), 256, 0, stream>>>(
        dwo1, pwin1, (long long)16384 * 128);
    gemm_bf16_wmma<<<dim3(2, 256, 1), 128, 0, stream>>>(
        pwin1, pw1m, pw1out, 16384, 128, 128, 128, 128, 128,
        0, 0, 0, nullptr, 0, pb1, 0);
    k_gate<<<16384, 128, 0, stream>>>(c1raw, pw1out, g1);

    // ---- gated block 2 (input concat[g1, x], K split into two 1152 passes) ----
    k_im2col<<<cdiv((long long)16384 * 1152, 256), 256, 0, stream>>>(g1, col);
    gemm_bf16_wmma<<<dim3(2, 256, 1), 128, 0, stream>>>(
        col, w2am, c2raw, 16384, 128, 1152, 1152, 1152, 128,
        0, 0, 0, nullptr, 0, b2, 0);
    k_im2col<<<cdiv((long long)16384 * 1152, 256), 256, 0, stream>>>(x, col);
    gemm_bf16_wmma<<<dim3(2, 256, 1), 128, 0, stream>>>(
        col, w2bm, c2raw, 16384, 128, 1152, 1152, 1152, 128,
        0, 0, 0, nullptr, 0, nullptr, 1);
    k_depthwise<<<cdiv((long long)4 * 4096 * 256, 256), 256, 0, stream>>>(
        g1, x, 128, 256, dw2, dwo2);
    k_cvt_bf16<<<cdiv((long long)16384 * 256, 256), 256, 0, stream>>>(
        dwo2, pwin2, (long long)16384 * 256);
    gemm_bf16_wmma<<<dim3(2, 256, 1), 128, 0, stream>>>(
        pwin2, pw2m, pw2out, 16384, 128, 256, 256, 256, 128,
        0, 0, 0, nullptr, 0, pb2, 0);
    k_gate<<<16384, 128, 0, stream>>>(c2raw, pw2out, out_f);
}
